// BinaryMLP_91276644974884
// MI455X (gfx1250) — 16-bit and 8-bit matrix operand order repaired and checked by reading, compiled, not run on hardware
//
#include <hip/hip_runtime.h>

typedef __attribute__((ext_vector_type(16))) __bf16 v16bf;
typedef __attribute__((ext_vector_type(8)))  float  v8f;
typedef __attribute__((ext_vector_type(8)))  int    v8i;

static constexpr int Bsz = 32768;
static constexpr int Din = 784;
static constexpr int H1  = 4096;
static constexpr int H2  = 4096;
static constexpr int NC  = 10;

__device__ __forceinline__ float bfround(float f) {
    unsigned u = __builtin_bit_cast(unsigned, f);
    unsigned r = (u + 0x7FFFu + ((u >> 16) & 1u)) & 0xFFFF0000u;
    return __builtin_bit_cast(float, r);
}
__device__ __forceinline__ __bf16 to_bf(float f_already_bf) {
    unsigned u = __builtin_bit_cast(unsigned, f_already_bf);
    unsigned short h = (unsigned short)(u >> 16);
    return __builtin_bit_cast(__bf16, h);
}

__global__ __launch_bounds__(256) void fc1_wmma(
    const float* __restrict__ x, const float* __restrict__ W1,
    const float* __restrict__ b1, const float* __restrict__ a1p,
    float* __restrict__ p1)
{
    __shared__ __bf16 Ah[128][40];
    __shared__ __bf16 Al[128][40];
    __shared__ __bf16 Wh[128][40];
    __shared__ __bf16 Wl[128][40];

    const int t    = threadIdx.x;
    const int m0   = blockIdx.y * 128;
    const int n0   = blockIdx.x * 128;
    const int lane = t & 31, wave = t >> 5;
    const int wm   = wave & 1, wn = wave >> 1;
    const int lm   = lane & 15, hh = lane >> 4;

    v8f acc[4][2];
    v8f zf = {};
    #pragma unroll
    for (int i = 0; i < 4; i++)
        #pragma unroll
        for (int j = 0; j < 2; j++) acc[i][j] = zf;

    const int row = t >> 1;
    const int cb  = (t & 1) * 16;

    for (int k0 = 0; k0 < Din; k0 += 32) {
        {
            const float* xs = x  + (size_t)(m0 + row) * Din + k0 + cb;
            const float* ws = W1 + (size_t)(n0 + row) * Din + k0 + cb;
            float xa[16], wa[16];
            if (k0 + cb + 16 <= Din) {
                const float4* x4 = (const float4*)xs;
                const float4* w4 = (const float4*)ws;
                #pragma unroll
                for (int q = 0; q < 4; q++) {
                    float4 v = x4[q];
                    xa[4*q+0]=v.x; xa[4*q+1]=v.y; xa[4*q+2]=v.z; xa[4*q+3]=v.w;
                    float4 w = w4[q];
                    wa[4*q+0]=w.x; wa[4*q+1]=w.y; wa[4*q+2]=w.z; wa[4*q+3]=w.w;
                }
            } else {
                #pragma unroll
                for (int i = 0; i < 16; i++) {
                    bool ok = (k0 + cb + i) < Din;
                    float xv = 0.f, wv = 0.f;
                    if (ok) { xv = xs[i]; wv = ws[i]; }
                    xa[i] = xv; wa[i] = wv;
                }
            }
            #pragma unroll
            for (int i = 0; i < 16; i++) {
                float hf = bfround(xa[i]);
                Ah[row][cb+i] = to_bf(hf);
                Al[row][cb+i] = to_bf(bfround(xa[i] - hf));
                float wf = bfround(wa[i]);
                Wh[row][cb+i] = to_bf(wf);
                Wl[row][cb+i] = to_bf(bfround(wa[i] - wf));
            }
        }
        __syncthreads();

        v16bf bh[2], bl[2];
        #pragma unroll
        for (int tn = 0; tn < 2; tn++) {
            int n = wn * 32 + tn * 16 + lm;
            #pragma unroll
            for (int i = 0; i < 16; i++) {
                bh[tn][i] = Wh[n][hh*16 + i];
                bl[tn][i] = Wl[n][hh*16 + i];
            }
        }
        #pragma unroll
        for (int tm = 0; tm < 4; tm++) {
            int m = wm * 64 + tm * 16 + lm;
            v16bf ah, al;
            #pragma unroll
            for (int i = 0; i < 16; i++) {
                ah[i] = Ah[m][hh*16 + i];
                al[i] = Al[m][hh*16 + i];
            }
            #pragma unroll
            for (int tn = 0; tn < 2; tn++) {
                acc[tm][tn] = __builtin_amdgcn_wmma_f32_16x16x32_bf16(
                    false, ah, false, bh[tn], (short)0, acc[tm][tn], false, false);
                acc[tm][tn] = __builtin_amdgcn_wmma_f32_16x16x32_bf16(
                    false, ah, false, bl[tn], (short)0, acc[tm][tn], false, false);
                acc[tm][tn] = __builtin_amdgcn_wmma_f32_16x16x32_bf16(
                    false, al, false, bh[tn], (short)0, acc[tm][tn], false, false);
            }
        }
        __syncthreads();
    }

    const float a1 = a1p[0];
    #pragma unroll
    for (int tn = 0; tn < 2; tn++) {
        int gn = n0 + wn * 32 + tn * 16 + lm;
        float bv = b1[gn];
        #pragma unroll
        for (int tm = 0; tm < 4; tm++) {
            #pragma unroll
            for (int r = 0; r < 8; r++) {
                int gm = m0 + wm * 64 + tm * 16 + r + 8 * hh;
                float v = acc[tm][tn][r] + bv;
                v = v > 0.f ? v : a1 * v;
                p1[(size_t)gm * H1 + gn] = v;
            }
        }
    }
}

__global__ __launch_bounds__(256) void colstats(
    const float* __restrict__ p, const float* __restrict__ g,
    const float* __restrict__ beta, float* __restrict__ scale,
    float* __restrict__ shift)
{
    int col = blockIdx.x * 256 + threadIdx.x;
    double s = 0.0, s2 = 0.0;
    for (int r = 0; r < Bsz; r++) {
        float v = p[(size_t)r * H1 + col];
        s  += v;
        s2 += (double)v * (double)v;
    }
    double mean = s / (double)Bsz;
    double var  = s2 / (double)Bsz - mean * mean;
    float sc = g[col] * rsqrtf((float)var + 1e-5f);
    scale[col] = sc;
    shift[col] = beta[col] - (float)mean * sc;
}

__global__ __launch_bounds__(256) void sign_act(
    const float* __restrict__ p, const float* __restrict__ scale,
    const float* __restrict__ shift, signed char* __restrict__ s)
{
    size_t i = ((size_t)blockIdx.x * 256 + threadIdx.x) * 4;
    int col = (int)(i & (size_t)(H1 - 1));
    float4 v = *(const float4*)(p + i);
    char4 o;
    o.x = (v.x * scale[col+0] + shift[col+0]) >= 0.f ? 1 : -1;
    o.y = (v.y * scale[col+1] + shift[col+1]) >= 0.f ? 1 : -1;
    o.z = (v.z * scale[col+2] + shift[col+2]) >= 0.f ? 1 : -1;
    o.w = (v.w * scale[col+3] + shift[col+3]) >= 0.f ? 1 : -1;
    *(char4*)(s + i) = o;
}

__global__ __launch_bounds__(256) void sign_w(
    const float* __restrict__ W, signed char* __restrict__ s)
{
    size_t i = ((size_t)blockIdx.x * 256 + threadIdx.x) * 4;
    float4 v = *(const float4*)(W + i);
    char4 o;
    o.x = v.x >= 0.f ? 1 : -1;
    o.y = v.y >= 0.f ? 1 : -1;
    o.z = v.z >= 0.f ? 1 : -1;
    o.w = v.w >= 0.f ? 1 : -1;
    *(char4*)(s + i) = o;
}

__global__ __launch_bounds__(256) void fc2_wmma(
    const signed char* __restrict__ s1, const signed char* __restrict__ sW2,
    const float* __restrict__ a2p, float* __restrict__ p2)
{
    __shared__ int As[2][128][20];
    __shared__ int Ws[2][128][20];

    const int t    = threadIdx.x;
    const int m0   = blockIdx.y * 128;
    const int n0   = blockIdx.x * 128;
    const int lane = t & 31, wave = t >> 5;
    const int wm   = wave & 1, wn = wave >> 1;
    const int lm   = lane & 15, hh = lane >> 4;

    v8i acc[4][2];
    v8i zi = {};
    #pragma unroll
    for (int i = 0; i < 4; i++)
        #pragma unroll
        for (int j = 0; j < 2; j++) acc[i][j] = zi;

    const int row = t >> 1;
    const int cbB = (t & 1) * 32;
    const int cwI = (t & 1) * 8;

    const signed char* ga = s1  + (size_t)(m0 + row) * H1 + cbB;
    const signed char* gb = sW2 + (size_t)(n0 + row) * H1 + cbB;

    auto issue_tile = [&](int buf, int k0) {
        unsigned la = (unsigned)(uintptr_t)&As[buf][row][cwI];
        unsigned lb = (unsigned)(uintptr_t)&Ws[buf][row][cwI];
        unsigned long long aa = (unsigned long long)(uintptr_t)(ga + k0);
        unsigned long long ab = (unsigned long long)(uintptr_t)(gb + k0);
        asm volatile(
            "global_load_async_to_lds_b128 %0, %1, off\n\t"
            "global_load_async_to_lds_b128 %0, %1, off offset:16\n\t"
            "global_load_async_to_lds_b128 %2, %3, off\n\t"
            "global_load_async_to_lds_b128 %2, %3, off offset:16"
            :: "v"(la), "v"(aa), "v"(lb), "v"(ab)
            : "memory");
    };

    issue_tile(0, 0);

    for (int k0 = 0; k0 < H1; k0 += 64) {
        const int buf = (k0 >> 6) & 1;
        if (k0 + 64 < H1) {
            issue_tile(buf ^ 1, k0 + 64);
            asm volatile("s_wait_asynccnt 0x4" ::: "memory");
        } else {
            asm volatile("s_wait_asynccnt 0x0" ::: "memory");
        }
        __syncthreads();

        v8i bf[2];
        #pragma unroll
        for (int tn = 0; tn < 2; tn++) {
            int n = wn * 32 + tn * 16 + lm;
            #pragma unroll
            for (int i = 0; i < 4; i++) {
                bf[tn][i]   = Ws[buf][n][hh*4 + i];
                bf[tn][4+i] = Ws[buf][n][8 + hh*4 + i];
            }
        }
        #pragma unroll
        for (int tm = 0; tm < 4; tm++) {
            int m = wm * 64 + tm * 16 + lm;
            v8i af;
            #pragma unroll
            for (int c = 0; c < 4; c++) {
                int idx = hh * 4 + (c & 1) * 2 + (c >> 1) * 8;
                af[2*c]   = As[buf][m][idx];
                af[2*c+1] = As[buf][m][idx + 1];
            }
            #pragma unroll
            for (int tn = 0; tn < 2; tn++) {
                acc[tm][tn] = __builtin_amdgcn_wmma_i32_16x16x64_iu8(
                    true, af, true, bf[tn], acc[tm][tn], false, false);
            }
        }
        __syncthreads();
    }

    const float a2 = a2p[0];
    #pragma unroll
    for (int tn = 0; tn < 2; tn++) {
        int gn = n0 + wn * 32 + tn * 16 + lm;
        #pragma unroll
        for (int tm = 0; tm < 4; tm++) {
            #pragma unroll
            for (int r = 0; r < 8; r++) {
                int gm = m0 + wm * 64 + tm * 16 + r + 8 * hh;
                float v = (float)acc[tm][tn][r];
                v = v > 0.f ? v : a2 * v;
                p2[(size_t)gm * H2 + gn] = v;
            }
        }
    }
}

__global__ __launch_bounds__(256) void fc3_lsm(
    const float* __restrict__ p2, const float* __restrict__ scale,
    const float* __restrict__ shift, const float* __restrict__ W3,
    const float* __restrict__ b3, float* __restrict__ out)
{
    int wave = threadIdx.x >> 5;
    int lane = threadIdx.x & 31;
    int rowid = blockIdx.x * 8 + wave;
    const float* hrow = p2 + (size_t)rowid * H2;

    float acc[NC];
    #pragma unroll
    for (int c = 0; c < NC; c++) acc[c] = 0.f;

    for (int j = lane; j < H2; j += 32) {
        float hv = hrow[j] * scale[j] + shift[j];
        #pragma unroll
        for (int c = 0; c < NC; c++) acc[c] += hv * W3[(size_t)c * H2 + j];
    }
    #pragma unroll
    for (int c = 0; c < NC; c++)
        #pragma unroll
        for (int m = 16; m >= 1; m >>= 1) acc[c] += __shfl_xor(acc[c], m, 32);

    if (lane == 0) {
        float l[NC], mx = -1e30f;
        #pragma unroll
        for (int c = 0; c < NC; c++) { l[c] = acc[c] + b3[c]; mx = fmaxf(mx, l[c]); }
        float sum = 0.f;
        #pragma unroll
        for (int c = 0; c < NC; c++) sum += expf(l[c] - mx);
        float lse = mx + logf(sum);
        #pragma unroll
        for (int c = 0; c < NC; c++) out[(size_t)rowid * NC + c] = l[c] - lse;
    }
}

extern "C" void kernel_launch(void* const* d_in, const int* in_sizes, int n_in,
                              void* d_out, int out_size, void* d_ws, size_t ws_size,
                              hipStream_t stream)
{
    const float* x   = (const float*)d_in[0];
    const float* W1  = (const float*)d_in[1];
    const float* b1  = (const float*)d_in[2];
    const float* a1  = (const float*)d_in[3];
    const float* g1  = (const float*)d_in[4];
    const float* bt1 = (const float*)d_in[5];
    const float* W2  = (const float*)d_in[6];
    const float* a2  = (const float*)d_in[7];
    const float* g2  = (const float*)d_in[8];
    const float* bt2 = (const float*)d_in[9];
    const float* W3  = (const float*)d_in[10];
    const float* b3  = (const float*)d_in[11];
    float* out = (float*)d_out;

    char* ws = (char*)d_ws;
    float*       p1     = (float*)ws;
    signed char* s1     = (signed char*)(ws + (size_t)Bsz * H1 * 4);
    signed char* sW2    = s1 + (size_t)Bsz * H1;
    float*       scale1 = (float*)(sW2 + (size_t)H2 * H1);
    float*       shift1 = scale1 + H1;
    float*       scale2 = shift1 + H1;
    float*       shift2 = scale2 + H2;
    float*       p2     = p1;

    fc1_wmma<<<dim3(H1/128, Bsz/128), 256, 0, stream>>>(x, W1, b1, a1, p1);
    colstats<<<H1/256, 256, 0, stream>>>(p1, g1, bt1, scale1, shift1);
    sign_act<<<(Bsz/4) * (H1/256), 256, 0, stream>>>(p1, scale1, shift1, s1);
    sign_w<<<(H2/4) * (H1/256), 256, 0, stream>>>(W2, sW2);
    fc2_wmma<<<dim3(H2/128, Bsz/128), 256, 0, stream>>>(s1, sW2, a2, p2);
    colstats<<<H2/256, 256, 0, stream>>>(p2, g2, bt2, scale2, shift2);
    fc3_lsm<<<Bsz/8, 256, 0, stream>>>(p2, scale2, shift2, W3, b3, out);
}
